// GNNRegressor_35046933135419
// MI455X (gfx1250) — matrix operand order repaired and checked by reading, not compiled, not run on hardware
//
#include <hip/hip_runtime.h>
#include <hip/hip_bf16.h>

// ---------------------------------------------------------------------------
// GNN regressor forward on gfx1250 (CDNA5, wave32, WMMA).
//   N=100000 nodes, E=1600000 edges, G=256 graphs, IN=64, HID=128, 3 layers.
// GEMMs: v_wmma_f32_16x16x32_f16, A tiles staged f32->f16 in LDS once per
// block (64 rows), B fragments pre-swizzled in workspace and held in VGPRs
// across 4 row tiles. All mode flags (KTN, COLS, DUAL, RELU, GATE) are
// template params: constant fragment indices, immediate store offsets,
// branch-free epilogues. Gate head fused into the gw1 GEMM via LDS atomics.
// ---------------------------------------------------------------------------

typedef __attribute__((ext_vector_type(16))) _Float16 v16h;
typedef __attribute__((ext_vector_type(8)))  float    v8f;

#define N_NODES  100000
#define N_EDGES  1600000
#define N_GRAPHS 256
#define IN_DIM   64
#define HID      128

// ------------------------- small utility kernels ---------------------------

__global__ __launch_bounds__(256) void fill_f32(float* p, float v, size_t n) {
    size_t i = (size_t)blockIdx.x * 256 + threadIdx.x;
    if (i < n) p[i] = v;
}

__global__ __launch_bounds__(256) void fill_u32(unsigned* p, unsigned v, int n) {
    int i = blockIdx.x * 256 + threadIdx.x;
    if (i < n) p[i] = v;
}

// monotone float <-> uint key (for atomicMax-based segment max over floats)
__device__ __forceinline__ unsigned fkey(float f) {
    unsigned u = __float_as_uint(f);
    return (u & 0x80000000u) ? ~u : (u | 0x80000000u);
}
__device__ __forceinline__ float funkey(unsigned k) {
    unsigned u = (k & 0x80000000u) ? (k & 0x7fffffffu) : ~k;
    return __uint_as_float(u);
}

// ----------------- weight conversion: f32 [K x C] -> f16 B-fragments -------
// Fragment layout: [kt][ct][lane(32)][h(16)] halves, contiguous.
//   N = ct*16 + (lane & 15)
//   K = kt*32 + (h%8) + 8*(lane >> 4) + 16*(h/8)
__global__ __launch_bounds__(256)
void conv_wfrag(const float* __restrict__ W, _Float16* __restrict__ F, int K, int C) {
    int tid = blockIdx.x * 256 + threadIdx.x;
    int ctiles = C >> 4;
    int total  = (K >> 5) * ctiles * 512;
    if (tid >= total) return;
    int hh   = tid & 15;
    int lane = (tid >> 4) & 31;
    int frag = tid >> 9;
    int ct   = frag % ctiles;
    int kt   = frag / ctiles;
    int n = (ct << 4) + (lane & 15);
    int k = (kt << 5) + (hh & 7) + ((lane >> 4) << 3) + ((hh >> 3) << 4);
    F[tid] = (_Float16)W[(size_t)k * C + n];
}

// -------------------- A-tile staging: global f32 -> LDS f16 fragments ------
// LDS layout: slot = ((rt*KTN + kt)*32 + lane), 16 halves per slot.
// A-fragment element mapping (16-bit A 16x32 per ISA):
//   M = lane & 15,  K = kt*32 + (h%8) + 8*(lane>>4) + 16*(h/8)
template <int KTN>
__device__ __forceinline__ void stage_tile(const float* __restrict__ A,
                                           int blockRow, int nRows,
                                           _Float16* __restrict__ sA) {
    constexpr int K = KTN * 32;
    constexpr int NSLOTS = 4 * KTN * 32;          // 4 row tiles
#pragma unroll
    for (int s0 = 0; s0 < NSLOTS; s0 += 256) {
        int slot = s0 + threadIdx.x;
        int lane = slot & 31;
        int ktrt = slot >> 5;
        int kt   = ktrt & (KTN - 1);
        int rt   = ktrt / KTN;
        if (blockRow + (rt << 4) >= nRows) continue;
        int row = blockRow + (rt << 4) + (lane & 15);
        const float* p = A + (size_t)row * K + (kt << 5) + ((lane >> 4) << 3);
        __builtin_prefetch((const void*)(p + (size_t)64 * K), 0, 1);
        float4 f0 = *(const float4*)(p);
        float4 f1 = *(const float4*)(p + 4);
        float4 f2 = *(const float4*)(p + 16);
        float4 f3 = *(const float4*)(p + 20);
        v16h a;
        a[0]  = (_Float16)f0.x; a[1]  = (_Float16)f0.y; a[2]  = (_Float16)f0.z; a[3]  = (_Float16)f0.w;
        a[4]  = (_Float16)f1.x; a[5]  = (_Float16)f1.y; a[6]  = (_Float16)f1.z; a[7]  = (_Float16)f1.w;
        a[8]  = (_Float16)f2.x; a[9]  = (_Float16)f2.y; a[10] = (_Float16)f2.z; a[11] = (_Float16)f2.w;
        a[12] = (_Float16)f3.x; a[13] = (_Float16)f3.y; a[14] = (_Float16)f3.z; a[15] = (_Float16)f3.w;
        *(v16h*)(sA + slot * 16) = a;
    }
}

// ----------------------- WMMA GEMM: Out = act(A*W [+ A2*W2] + bias) --------
// Block = 256 threads = 8 waves; wave ct covers one 16-col tile; block covers
// 64 rows (4 row tiles per wave, B fragments reused in VGPRs). GATE mode:
// instead of storing, gateOut[row] = sum_col(relu(...) * gateW[col]) + gateB[0]
// accumulated via LDS ds_add_f32.
template <int KTN, int COLS, bool DUAL, bool RELU, bool GATE>
__global__ __launch_bounds__(256)
void gemm_wmma(const float* __restrict__ A, const _Float16* __restrict__ Wf,
               const float* __restrict__ bias,
               const float* __restrict__ A2, const _Float16* __restrict__ W2f,
               float* __restrict__ Out, int nRows,
               const float* __restrict__ gateW, const float* __restrict__ gateB,
               float* __restrict__ gateOut) {
    constexpr int CTILES = COLS >> 4;
    __shared__ _Float16 sA[4 * KTN * 32 * 16];
    __shared__ _Float16 sA2[DUAL ? 4 * KTN * 32 * 16 : 16];
    __shared__ float    sGate[GATE ? 64 : 1];

    const int lane     = threadIdx.x & 31;
    const int wave     = threadIdx.x >> 5;
    const int blockRow = blockIdx.x << 6;         // 64 rows per block

    stage_tile<KTN>(A, blockRow, nRows, sA);
    if (DUAL) stage_tile<KTN>(A2, blockRow, nRows, sA2);
    if (GATE && threadIdx.x < 64) sGate[threadIdx.x] = 0.0f;
    __syncthreads();

    if (wave < CTILES) {
        const int g = lane >> 4;
        const int m = lane & 15;
        const int col = (wave << 4) + m;

        // B fragments held in registers across the 4 row tiles
        v16h bf[KTN];
        v16h b2f[DUAL ? KTN : 1];
#pragma unroll
        for (int kt = 0; kt < KTN; ++kt) {
            bf[kt] = *(const v16h*)(Wf + (((kt * CTILES + wave) << 5) + lane) * 16);
            if (DUAL)
                b2f[kt] = *(const v16h*)(W2f + (((kt * CTILES + wave) << 5) + lane) * 16);
        }
        const float bv = bias[col];
        const float gw = GATE ? gateW[col] : 0.0f;

        for (int rt = 0; rt < 4; ++rt) {
            if (blockRow + (rt << 4) >= nRows) break;   // block-uniform tail
            v8f acc = {};
#pragma unroll
            for (int kt = 0; kt < KTN; ++kt) {
                v16h a = *(const v16h*)(sA + ((rt * KTN + kt) * 32 + lane) * 16);
                acc = __builtin_amdgcn_wmma_f32_16x16x32_f16(
                        false, a, false, bf[kt], (short)0, acc, false, false);
            }
            if (DUAL) {
#pragma unroll
                for (int kt = 0; kt < KTN; ++kt) {
                    v16h a = *(const v16h*)(sA2 + ((rt * KTN + kt) * 32 + lane) * 16);
                    acc = __builtin_amdgcn_wmma_f32_16x16x32_f16(
                            false, a, false, b2f[kt], (short)0, acc, false, false);
                }
            }
            // epilogue: M = r + 8*g within tile, N = lane & 15
            if (!GATE) {
                float* orow = Out + (size_t)(blockRow + (rt << 4) + (g << 3)) * COLS + col;
#pragma unroll
                for (int r = 0; r < 8; ++r) {
                    float v = acc[r] + bv;
                    if (RELU) v = fmaxf(v, 0.0f);
                    orow[r * COLS] = v;                 // constant immediate offsets
                }
            } else {
#pragma unroll
                for (int r = 0; r < 8; ++r) {
                    float v = fmaxf(acc[r] + bv, 0.0f);
                    atomicAdd(&sGate[(rt << 4) + (g << 3) + r], v * gw);
                }
            }
        }
    }

    if (GATE) {
        __syncthreads();
        int row = blockRow + threadIdx.x;
        if (threadIdx.x < 64 && row < nRows)
            gateOut[row] = sGate[threadIdx.x] + gateB[0];
    }
}

// ----------------------------- edge kernels --------------------------------

__global__ __launch_bounds__(256)
void count_deg(const int* __restrict__ dst, float* __restrict__ cnt, int nE) {
    int e = blockIdx.x * 256 + threadIdx.x;
    if (e < nE) atomicAdd(&cnt[dst[e]], 1.0f);
}

// msg[dst] += h[src]; one wave per edge, 4 columns per lane (float4 row load)
__global__ __launch_bounds__(256)
void scatter_add(const int* __restrict__ src, const int* __restrict__ dst,
                 const float* __restrict__ Hn, float* __restrict__ Msg, int nE) {
    int e = (blockIdx.x << 3) + (threadIdx.x >> 5);
    if (e >= nE) return;
    int lane = threadIdx.x & 31;
    int s = src[e], d = dst[e];
    const float4 v = *(const float4*)(Hn + (size_t)s * HID + (lane << 2));
    float* mp = Msg + (size_t)d * HID + (lane << 2);
    atomicAdd(mp + 0, v.x);
    atomicAdd(mp + 1, v.y);
    atomicAdd(mp + 2, v.z);
    atomicAdd(mp + 3, v.w);
}

// agg = count>0 ? msg / max(count,1) : 0   (in place)
__global__ __launch_bounds__(256)
void norm_mean(float* __restrict__ Msg, const float* __restrict__ cnt, size_t total) {
    size_t i = (size_t)blockIdx.x * 256 + threadIdx.x;
    if (i >= total) return;
    float c = cnt[i >> 7];
    float v = Msg[i];
    Msg[i] = (c > 0.0f) ? v / fmaxf(c, 1.0f) : 0.0f;
}

// ---------------------- gate / attention pooling ---------------------------

// out[row] = dot(A[row, 0:K], w) + bias[0]; one wave per row, shfl_xor reduce
__global__ __launch_bounds__(256)
void dot_rows(const float* __restrict__ A, const float* __restrict__ w,
              const float* __restrict__ bias, float* __restrict__ out, int n, int K) {
    int row = (blockIdx.x << 3) + (threadIdx.x >> 5);
    if (row >= n) return;
    int lane = threadIdx.x & 31;
    float s = 0.0f;
    for (int c = lane; c < K; c += 32) s += A[(size_t)row * K + c] * w[c];
#pragma unroll
    for (int off = 16; off > 0; off >>= 1) s += __shfl_xor(s, off, 32);
    if (lane == 0) out[row] = s + bias[0];
}

__global__ __launch_bounds__(256)
void seg_max_gate(const float* __restrict__ gate, const int* __restrict__ batch,
                  unsigned* __restrict__ gkey, int n) {
    int i = blockIdx.x * 256 + threadIdx.x;
    if (i < n) atomicMax(&gkey[batch[i]], fkey(gate[i]));
}

__global__ __launch_bounds__(256)
void softmax_e(const float* __restrict__ gate, const int* __restrict__ batch,
               const unsigned* __restrict__ gkey, float* __restrict__ eexp,
               float* __restrict__ denom, int n) {
    int i = blockIdx.x * 256 + threadIdx.x;
    if (i >= n) return;
    int b = batch[i];
    float e = __expf(gate[i] - funkey(gkey[b]));
    eexp[i] = e;
    atomicAdd(&denom[b], e);
}

// pooled[b] += (e/denom[b]) * h[i]; one wave per node, 4 cols/lane
__global__ __launch_bounds__(256)
void pool_attn(const float* __restrict__ Hn, const float* __restrict__ eexp,
               const float* __restrict__ denom, const int* __restrict__ batch,
               float* __restrict__ pooled, int n) {
    int i = (blockIdx.x << 3) + (threadIdx.x >> 5);
    if (i >= n) return;
    int lane = threadIdx.x & 31;
    int b = batch[i];
    float alpha = eexp[i] / denom[b];
    const float4 v = *(const float4*)(Hn + (size_t)i * HID + (lane << 2));
    float* pp = pooled + (size_t)b * HID + (lane << 2);
    atomicAdd(pp + 0, alpha * v.x);
    atomicAdd(pp + 1, alpha * v.y);
    atomicAdd(pp + 2, alpha * v.z);
    atomicAdd(pp + 3, alpha * v.w);
}

// ------------------------------- driver ------------------------------------

static inline int cdiv(size_t a, size_t b) { return (int)((a + b - 1) / b); }

extern "C" void kernel_launch(void* const* d_in, const int* in_sizes, int n_in,
                              void* d_out, int out_size, void* d_ws, size_t ws_size,
                              hipStream_t stream) {
    (void)in_sizes; (void)n_in; (void)out_size; (void)ws_size;

    // ---- inputs (setup_inputs dict order, params dict flattened in order) ----
    const float* x     = (const float*)d_in[0];   // [N, 64]
    const int*   eidx  = (const int*)d_in[1];     // [2, E]
    const int*   batch = (const int*)d_in[2];     // [N]
    const float* w_in  = (const float*)d_in[3];   // [64,128]
    const float* b_in  = (const float*)d_in[4];   // [128]
    const float* w_l[3] = { (const float*)d_in[5],  (const float*)d_in[8],  (const float*)d_in[11] };
    const float* b_l[3] = { (const float*)d_in[6],  (const float*)d_in[9],  (const float*)d_in[12] };
    const float* w_r[3] = { (const float*)d_in[7],  (const float*)d_in[10], (const float*)d_in[13] };
    const float* gw1 = (const float*)d_in[14];    // [128,128]
    const float* gb1 = (const float*)d_in[15];
    const float* gw2 = (const float*)d_in[16];    // [128,1]
    const float* gb2 = (const float*)d_in[17];
    const float* hw1 = (const float*)d_in[18];    // [128,128]
    const float* hb1 = (const float*)d_in[19];
    const float* hw2 = (const float*)d_in[20];    // [128,64]
    const float* hb2 = (const float*)d_in[21];
    const float* hw3 = (const float*)d_in[22];    // [64,1]
    const float* hb3 = (const float*)d_in[23];
    float* out = (float*)d_out;                   // [256]

    const int* src = eidx;            // edge_index[0]
    const int* dst = eidx + N_EDGES;  // edge_index[1]

    // ---- workspace carve-out (256B aligned regions) ----
    char* ws = (char*)d_ws;
    size_t off = 0;
    auto carve = [&](size_t bytes) -> char* {
        char* p = ws + off;
        off = (off + bytes + 255) & ~(size_t)255;
        return p;
    };
    float*    hA     = (float*)carve((size_t)N_NODES * HID * 4);
    float*    hB     = (float*)carve((size_t)N_NODES * HID * 4);
    float*    msg    = (float*)carve((size_t)N_NODES * HID * 4);
    float*    cnt    = (float*)carve((size_t)N_NODES * 4);
    float*    gate   = (float*)carve((size_t)N_NODES * 4);
    float*    eexp   = (float*)carve((size_t)N_NODES * 4);
    unsigned* gkey   = (unsigned*)carve((size_t)N_GRAPHS * 4);
    float*    denom  = (float*)carve((size_t)N_GRAPHS * 4);
    float*    pooled = (float*)carve((size_t)N_GRAPHS * HID * 4);
    float*    z1     = (float*)carve((size_t)N_GRAPHS * HID * 4);
    float*    z2     = (float*)carve((size_t)N_GRAPHS * 64 * 4);
    _Float16* f_win  = (_Float16*)carve((size_t)2 * 8 * 512 * 2);   // K=64,  C=128
    _Float16* f_wl[3], *f_wr[3];
    for (int i = 0; i < 3; ++i) {
        f_wl[i] = (_Float16*)carve((size_t)4 * 8 * 512 * 2);        // K=128, C=128
        f_wr[i] = (_Float16*)carve((size_t)4 * 8 * 512 * 2);
    }
    _Float16* f_gw1 = (_Float16*)carve((size_t)4 * 8 * 512 * 2);
    _Float16* f_hw1 = (_Float16*)carve((size_t)4 * 8 * 512 * 2);
    _Float16* f_hw2 = (_Float16*)carve((size_t)4 * 4 * 512 * 2);    // K=128, C=64

    // ---- 0) convert weights to pre-swizzled f16 B-fragments ----
    conv_wfrag<<<cdiv(2 * 8 * 512, 256), 256, 0, stream>>>(w_in, f_win, 64, 128);
    for (int i = 0; i < 3; ++i) {
        conv_wfrag<<<cdiv(4 * 8 * 512, 256), 256, 0, stream>>>(w_l[i], f_wl[i], 128, 128);
        conv_wfrag<<<cdiv(4 * 8 * 512, 256), 256, 0, stream>>>(w_r[i], f_wr[i], 128, 128);
    }
    conv_wfrag<<<cdiv(4 * 8 * 512, 256), 256, 0, stream>>>(gw1, f_gw1, 128, 128);
    conv_wfrag<<<cdiv(4 * 8 * 512, 256), 256, 0, stream>>>(hw1, f_hw1, 128, 128);
    conv_wfrag<<<cdiv(4 * 8 * 512, 256), 256, 0, stream>>>(hw2, f_hw2, 128, 64);

    // ---- 1) in-degree counts ----
    fill_f32<<<cdiv(N_NODES, 256), 256, 0, stream>>>(cnt, 0.0f, N_NODES);
    count_deg<<<cdiv(N_EDGES, 256), 256, 0, stream>>>(dst, cnt, N_EDGES);

    // ---- 2) h = x @ w_in + b_in ----
    gemm_wmma<2, 128, false, false, false><<<cdiv(N_NODES, 64), 256, 0, stream>>>(
        x, f_win, b_in, nullptr, nullptr, hA, N_NODES, nullptr, nullptr, nullptr);

    // ---- 3) message-passing layers ----
    float* hcur = hA;
    float* hnew = hB;
    const size_t nh = (size_t)N_NODES * HID;
    for (int i = 0; i < 3; ++i) {
        fill_f32<<<cdiv(nh, 256), 256, 0, stream>>>(msg, 0.0f, nh);
        scatter_add<<<cdiv(N_EDGES, 8), 256, 0, stream>>>(src, dst, hcur, msg, N_EDGES);
        norm_mean<<<cdiv(nh, 256), 256, 0, stream>>>(msg, cnt, nh);
        // h = relu(agg @ w_l + b_l + h @ w_r)
        gemm_wmma<4, 128, true, true, false><<<cdiv(N_NODES, 64), 256, 0, stream>>>(
            msg, f_wl[i], b_l[i], hcur, f_wr[i], hnew, N_NODES, nullptr, nullptr, nullptr);
        float* t = hcur; hcur = hnew; hnew = t;
    }

    // ---- 4) gate = relu(h@gw1+gb1)@gw2 + gb2  (fused, no g1 materialized) --
    gemm_wmma<4, 128, false, true, true><<<cdiv(N_NODES, 64), 256, 0, stream>>>(
        hcur, f_gw1, gb1, nullptr, nullptr, nullptr, N_NODES, gw2, gb2, gate);

    // ---- 5) per-graph softmax + attention pooling ----
    fill_u32<<<1, 256, 0, stream>>>(gkey, 0x007FFFFFu /* fkey(-inf) */, N_GRAPHS);
    fill_f32<<<1, 256, 0, stream>>>(denom, 0.0f, N_GRAPHS);
    fill_f32<<<cdiv((size_t)N_GRAPHS * HID, 256), 256, 0, stream>>>(
        pooled, 0.0f, (size_t)N_GRAPHS * HID);
    seg_max_gate<<<cdiv(N_NODES, 256), 256, 0, stream>>>(gate, batch, gkey, N_NODES);
    softmax_e<<<cdiv(N_NODES, 256), 256, 0, stream>>>(gate, batch, gkey, eexp, denom, N_NODES);
    pool_attn<<<cdiv(N_NODES, 8), 256, 0, stream>>>(hcur, eexp, denom, batch, pooled, N_NODES);

    // ---- 6) head MLP ----
    gemm_wmma<4, 128, false, true, false><<<cdiv(N_GRAPHS, 64), 256, 0, stream>>>(
        pooled, f_hw1, hb1, nullptr, nullptr, z1, N_GRAPHS, nullptr, nullptr, nullptr);
    gemm_wmma<4, 64, false, true, false><<<cdiv(N_GRAPHS, 64), 256, 0, stream>>>(
        z1, f_hw2, hb2, nullptr, nullptr, z2, N_GRAPHS, nullptr, nullptr, nullptr);
    dot_rows<<<cdiv(N_GRAPHS, 8), 256, 0, stream>>>(z2, hw3, hb3, out, N_GRAPHS, 64);
}
